// SoftmaxAttention_88742614270726
// MI455X (gfx1250) — matrix operand order repaired and checked by reading, not compiled, not run on hardware
//
#include <hip/hip_runtime.h>

typedef __attribute__((ext_vector_type(16))) __bf16 v16bf;
typedef __attribute__((ext_vector_type(8)))  __bf16 v8bf;
typedef __attribute__((ext_vector_type(8)))  float  v8f;
typedef __attribute__((ext_vector_type(4)))  float  v4f;

constexpr int BATCH  = 32;
constexpr int L      = 192;    // Lp == Lh
constexpr int DMODEL = 1024;   // D == Dg
constexpr int ATT_N  = BATCH * L * L;        // one attention map
constexpr int TEN_N  = BATCH * L * DMODEL;   // one big tensor

// ---------------------------------------------------------------------------
// WMMA fragment loaders (CDNA5 16x16x32 bf16 layouts, ISA 7.12.2):
//  A (16x32, MxK): lane r(=lane&15), half hi(=lane>>4); elements
//    [0..7]  = row r, K = k0 + hi*8 .. +7
//    [8..15] = row r, K = k0 + 16 + hi*8 .. +7
//  B (32x16, KxN): lane r holds column N=r; same K order as A
// ---------------------------------------------------------------------------

__device__ inline v16bf load_a_f32(const float* base, int ld, int row0, int k0, int lane) {
    int r = lane & 15, hi = lane >> 4;
    const float* p0 = base + (size_t)(row0 + r) * ld + k0 + hi * 8;
    v4f x0 = *(const v4f*)(p0);
    v4f x1 = *(const v4f*)(p0 + 4);
    v4f x2 = *(const v4f*)(p0 + 16);
    v4f x3 = *(const v4f*)(p0 + 20);
    v16bf o;
#pragma unroll
    for (int i = 0; i < 4; ++i) {
        o[i]      = (__bf16)x0[i];
        o[4 + i]  = (__bf16)x1[i];
        o[8 + i]  = (__bf16)x2[i];
        o[12 + i] = (__bf16)x3[i];
    }
    return o;
}

__device__ inline v16bf load_b_f32(const float* base, int ld, int n0, int k0, int lane) {
    int r = lane & 15, hi = lane >> 4;
    const float* p = base + (size_t)(n0 + r) * ld + k0 + hi * 8;
    v4f x0 = *(const v4f*)(p);
    v4f x1 = *(const v4f*)(p + 4);
    v4f x2 = *(const v4f*)(p + 16);
    v4f x3 = *(const v4f*)(p + 20);
    v16bf o;
#pragma unroll
    for (int i = 0; i < 4; ++i) {
        o[i]      = (__bf16)x0[i];
        o[4 + i]  = (__bf16)x1[i];
        o[8 + i]  = (__bf16)x2[i];
        o[12 + i] = (__bf16)x3[i];
    }
    return o;
}

__device__ inline v16bf load_a_bf16(const __bf16* base, int ld, int row0, int k0, int lane) {
    int r = lane & 15, hi = lane >> 4;
    const __bf16* p0 = base + (size_t)(row0 + r) * ld + k0 + hi * 8;
    v8bf lo = *(const v8bf*)(p0);
    v8bf hh = *(const v8bf*)(p0 + 16);
    return __builtin_shufflevector(lo, hh, 0,1,2,3,4,5,6,7,8,9,10,11,12,13,14,15);
}

__device__ inline v16bf load_b_bf16(const __bf16* base, int ld, int n0, int k0, int lane) {
    int r = lane & 15, hi = lane >> 4;
    const __bf16* p = base + (size_t)(n0 + r) * ld + k0 + hi * 8;
    v8bf lo = *(const v8bf*)(p);
    v8bf hh = *(const v8bf*)(p + 16);
    return __builtin_shufflevector(lo, hh, 0,1,2,3,4,5,6,7,8,9,10,11,12,13,14,15);
}

// ---------------------------------------------------------------------------
// Kernel 1: per batch, per tensor: f32 [L x D] -> bf16 transposed [D x L]
// and (optionally) bf16 row-major [L x D].
// ---------------------------------------------------------------------------
__global__ __launch_bounds__(256) void transpose_cvt_kernel(
    const float* __restrict__ t0, const float* __restrict__ t1,
    const float* __restrict__ t2, const float* __restrict__ t3,
    __bf16* __restrict__ o0, __bf16* __restrict__ o1,
    __bf16* __restrict__ o2, __bf16* __restrict__ o3,
    __bf16* __restrict__ r0, __bf16* __restrict__ r1,
    __bf16* __restrict__ r2, __bf16* __restrict__ r3,
    int write_row)
{
    __shared__ float tile[32][33];
    int which = blockIdx.z >> 5;
    int b     = blockIdx.z & 31;
    const float* src = (which == 0) ? t0 : (which == 1) ? t1 : (which == 2) ? t2 : t3;
    __bf16*      dst = (which == 0) ? o0 : (which == 1) ? o1 : (which == 2) ? o2 : o3;
    __bf16*      row = (which == 0) ? r0 : (which == 1) ? r1 : (which == 2) ? r2 : r3;
    src += (size_t)b * L * DMODEL;
    dst += (size_t)b * DMODEL * L;
    row += (size_t)b * L * DMODEL;
    int d0 = blockIdx.x * 32, k0 = blockIdx.y * 32;
    int tx = threadIdx.x, ty = threadIdx.y;
#pragma unroll
    for (int i = 0; i < 4; ++i) {
        float v = src[(size_t)(k0 + ty + 8 * i) * DMODEL + d0 + tx];
        tile[ty + 8 * i][tx] = v;
        if (write_row)
            row[(size_t)(k0 + ty + 8 * i) * DMODEL + d0 + tx] = (__bf16)v;
    }
    __syncthreads();
#pragma unroll
    for (int i = 0; i < 4; ++i)
        dst[(size_t)(d0 + ty + 8 * i) * L + k0 + tx] = (__bf16)tile[tx][ty + 8 * i];
}

// ---------------------------------------------------------------------------
// Kernel 2a (fast path): S = P*H^T, Sg = Pg*Hg^T from pre-converted bf16.
// One wave computes a 16x64 output strip: 1 A fragment reused over 4 B
// fragments / 4 accumulators per K-step. K-loop over D=1024 step 32.
// ---------------------------------------------------------------------------
__global__ __launch_bounds__(128) void sim_bf16_kernel(
    const __bf16* __restrict__ Pb, const __bf16* __restrict__ Hb,
    const __bf16* __restrict__ Pgb, const __bf16* __restrict__ Hgb,
    float* __restrict__ S, float* __restrict__ Sg)
{
    int lane = threadIdx.x & 31;
    int id   = blockIdx.x * 4 + (threadIdx.x >> 5);   // 0 .. 2*BATCH*36-1
    int mat  = id / (BATCH * 36);
    int rem  = id % (BATCH * 36);
    int b    = rem / 36, t = rem % 36;
    int tp   = (t / 3) * 16;   // premise-row tile (12)
    int th0  = (t % 3) * 64;   // hypothesis-col strip (3 strips of 4 tiles)
    const __bf16* Abase = ((mat == 0) ? Pb : Pgb) + (size_t)b * L * DMODEL;
    const __bf16* Bbase = ((mat == 0) ? Hb : Hgb) + (size_t)b * L * DMODEL;
    float*        Out   = ((mat == 0) ? S : Sg) + (size_t)b * L * L;

    v8f c0 = {}, c1 = {}, c2 = {}, c3 = {};
#pragma unroll 2
    for (int k0 = 0; k0 < DMODEL; k0 += 32) {
        v16bf a  = load_a_bf16(Abase, DMODEL, tp, k0, lane);
        v16bf b0 = load_b_bf16(Bbase, DMODEL, th0,      k0, lane);
        v16bf b1 = load_b_bf16(Bbase, DMODEL, th0 + 16, k0, lane);
        v16bf b2 = load_b_bf16(Bbase, DMODEL, th0 + 32, k0, lane);
        v16bf b3 = load_b_bf16(Bbase, DMODEL, th0 + 48, k0, lane);
        c0 = __builtin_amdgcn_wmma_f32_16x16x32_bf16(false, a, false, b0, (short)0, c0, false, false);
        c1 = __builtin_amdgcn_wmma_f32_16x16x32_bf16(false, a, false, b1, (short)0, c1, false, false);
        c2 = __builtin_amdgcn_wmma_f32_16x16x32_bf16(false, a, false, b2, (short)0, c2, false, false);
        c3 = __builtin_amdgcn_wmma_f32_16x16x32_bf16(false, a, false, b3, (short)0, c3, false, false);
    }
    int n = lane & 15, mb = (lane >> 4) * 8;
#pragma unroll
    for (int v = 0; v < 8; ++v) {
        float* orow = Out + (size_t)(tp + mb + v) * L + th0 + n;
        orow[0]  = c0[v];
        orow[16] = c1[v];
        orow[32] = c2[v];
        orow[48] = c3[v];
    }
}

// ---------------------------------------------------------------------------
// Kernel 2b (fallback if workspace too small): on-the-fly f32->bf16 cvt.
// ---------------------------------------------------------------------------
__global__ __launch_bounds__(128) void sim_f32_kernel(
    const float* __restrict__ P,  const float* __restrict__ H,
    const float* __restrict__ Pg, const float* __restrict__ Hg,
    float* __restrict__ S, float* __restrict__ Sg)
{
    int lane = threadIdx.x & 31;
    int id   = blockIdx.x * 4 + (threadIdx.x >> 5);   // 0 .. 2*BATCH*144-1
    int mat  = id / (BATCH * 144);
    int rem  = id % (BATCH * 144);
    int b    = rem / 144, t = rem % 144;
    int tp   = (t / 12) * 16;
    int th   = (t % 12) * 16;
    const float* Abase = ((mat == 0) ? P : Pg) + (size_t)b * L * DMODEL;
    const float* Bbase = ((mat == 0) ? H : Hg) + (size_t)b * L * DMODEL;
    float*       Out   = ((mat == 0) ? S : Sg) + (size_t)b * L * L;

    v8f c = {};
#pragma unroll 4
    for (int k0 = 0; k0 < DMODEL; k0 += 32) {
        v16bf a  = load_a_f32(Abase, DMODEL, tp, k0, lane);
        v16bf bb = load_b_f32(Bbase, DMODEL, th, k0, lane);
        c = __builtin_amdgcn_wmma_f32_16x16x32_bf16(false, a, false, bb, (short)0, c, false, false);
    }
    int n = lane & 15, mb = (lane >> 4) * 8;
#pragma unroll
    for (int v = 0; v < 8; ++v)
        Out[(size_t)(tp + mb + v) * L + th + n] = c[v];
}

// ---------------------------------------------------------------------------
// Kernel 3: masked softmax over one row of a (possibly transposed) 192x192
// matrix. Mirrors reference: p=softmax(s*m); p*=m; p/=(sum(p)+eps).
// ---------------------------------------------------------------------------
__global__ __launch_bounds__(256) void masked_softmax_kernel(
    const float* __restrict__ Sb, int rs, int cs,
    const float* __restrict__ mask, float* __restrict__ out)
{
    __shared__ float red[256];
    int b = blockIdx.x / L;
    int i = blockIdx.x % L;
    int j = threadIdx.x;
    const float* Sp   = Sb + (size_t)b * L * L;
    const float* mrow = mask + b * L;

    float m = (j < L) ? mrow[j] : 0.f;
    float s = (j < L) ? Sp[(size_t)i * rs + (size_t)j * cs] * m : -3.0e38f;

    red[j] = s; __syncthreads();
#pragma unroll
    for (int st = 128; st > 0; st >>= 1) { if (j < st) red[j] = fmaxf(red[j], red[j + st]); __syncthreads(); }
    float mx = red[0]; __syncthreads();

    float e = (j < L) ? __expf(s - mx) : 0.f;
    red[j] = e; __syncthreads();
#pragma unroll
    for (int st = 128; st > 0; st >>= 1) { if (j < st) red[j] += red[j + st]; __syncthreads(); }
    float Z = red[0]; __syncthreads();

    float pm = (e / Z) * m;
    red[j] = (j < L) ? pm : 0.f; __syncthreads();
#pragma unroll
    for (int st = 128; st > 0; st >>= 1) { if (j < st) red[j] += red[j + st]; __syncthreads(); }
    float denom = red[0] + 1e-13f;

    if (j < L) out[(size_t)b * L * L + (size_t)i * L + j] = pm / denom;
}

// ---------------------------------------------------------------------------
// Kernel 4: tot = attn + attn_g (f32 outputs) + bf16 weight copies for GEMMs.
// ---------------------------------------------------------------------------
__global__ __launch_bounds__(256) void combine_kernel(
    const float* __restrict__ ph,  const float* __restrict__ hp,
    const float* __restrict__ phg, const float* __restrict__ hpg,
    float* __restrict__ totp, float* __restrict__ toth,
    __bf16* __restrict__ wtotp, __bf16* __restrict__ wtoth,
    __bf16* __restrict__ wphg,  __bf16* __restrict__ whpg)
{
    int idx = blockIdx.x * 256 + threadIdx.x;
    if (idx >= ATT_N) return;
    float a = ph[idx], ag = phg[idx];
    float tp = a + ag;
    totp[idx]  = tp;
    wtotp[idx] = (__bf16)tp;
    wphg[idx]  = (__bf16)ag;
    float h = hp[idx], hg = hpg[idx];
    float th = h + hg;
    toth[idx]  = th;
    wtoth[idx] = (__bf16)th;
    whpg[idx]  = (__bf16)hg;
}

// ---------------------------------------------------------------------------
// Kernel 5: attended = rowmask * (W[LxL] @ T[LxD]); T supplied transposed
// (bf16 [D x L]) so the WMMA B operand is K-contiguous per lane.
// One wave computes a 16x64 output strip (4 accumulators), K = 192.
// ---------------------------------------------------------------------------
__global__ __launch_bounds__(128) void wsum_kernel(
    const __bf16* __restrict__ Wtotp, const __bf16* __restrict__ Wtoth,
    const __bf16* __restrict__ Wphg,  const __bf16* __restrict__ Whpg,
    const __bf16* __restrict__ TtH,   const __bf16* __restrict__ TtP,
    const __bf16* __restrict__ TtHg,  const __bf16* __restrict__ TtPg,
    const float*  __restrict__ maskP, const float* __restrict__ maskH,
    float* __restrict__ outAP,  float* __restrict__ outAH,
    float* __restrict__ outAPg, float* __restrict__ outAHg)
{
    int lane = threadIdx.x & 31;
    int id   = blockIdx.x * 4 + (threadIdx.x >> 5);   // 0 .. 4*BATCH*192-1
    int g    = id / (BATCH * 192);
    int rem  = id % (BATCH * 192);
    int b    = rem / 192, t = rem % 192;
    int q0   = (t / 16) * 16;   // 12 q-tiles
    int d0   = (t % 16) * 64;   // 16 strips of 4 d-tiles

    const __bf16* W; const __bf16* Tt; const float* mk; float* out;
    if (g == 0)      { W = Wtotp; Tt = TtH;  mk = maskP; out = outAP;  }
    else if (g == 1) { W = Wtoth; Tt = TtP;  mk = maskH; out = outAH;  }
    else if (g == 2) { W = Wphg;  Tt = TtHg; mk = maskP; out = outAPg; }
    else             { W = Whpg;  Tt = TtPg; mk = maskH; out = outAHg; }
    W   += (size_t)b * L * L;
    Tt  += (size_t)b * DMODEL * L;
    out += (size_t)b * L * DMODEL;
    mk  += b * L;

    v8f c0 = {}, c1 = {}, c2 = {}, c3 = {};
#pragma unroll
    for (int k0 = 0; k0 < L; k0 += 32) {
        v16bf a  = load_a_bf16(W,  L, q0, k0, lane);
        v16bf b0 = load_b_bf16(Tt, L, d0,      k0, lane);
        v16bf b1 = load_b_bf16(Tt, L, d0 + 16, k0, lane);
        v16bf b2 = load_b_bf16(Tt, L, d0 + 32, k0, lane);
        v16bf b3 = load_b_bf16(Tt, L, d0 + 48, k0, lane);
        c0 = __builtin_amdgcn_wmma_f32_16x16x32_bf16(false, a, false, b0, (short)0, c0, false, false);
        c1 = __builtin_amdgcn_wmma_f32_16x16x32_bf16(false, a, false, b1, (short)0, c1, false, false);
        c2 = __builtin_amdgcn_wmma_f32_16x16x32_bf16(false, a, false, b2, (short)0, c2, false, false);
        c3 = __builtin_amdgcn_wmma_f32_16x16x32_bf16(false, a, false, b3, (short)0, c3, false, false);
    }
    int n = lane & 15, mb = (lane >> 4) * 8;
#pragma unroll
    for (int v = 0; v < 8; ++v) {
        int q = q0 + mb + v;
        float rm = mk[q];
        float* orow = out + (size_t)q * DMODEL + d0 + n;
        orow[0]  = c0[v] * rm;
        orow[16] = c1[v] * rm;
        orow[32] = c2[v] * rm;
        orow[48] = c3[v] * rm;
    }
}

// ---------------------------------------------------------------------------
extern "C" void kernel_launch(void* const* d_in, const int* in_sizes, int n_in,
                              void* d_out, int out_size, void* d_ws, size_t ws_size,
                              hipStream_t stream) {
    (void)in_sizes; (void)n_in; (void)out_size;

    const float* P     = (const float*)d_in[0];   // premise_batch      [32,192,1024]
    const float* maskP = (const float*)d_in[1];   // premise_mask       [32,192]
    const float* H     = (const float*)d_in[2];   // hypothesis_batch   [32,192,1024]
    const float* maskH = (const float*)d_in[3];   // hypothesis_mask    [32,192]
    const float* Pg    = (const float*)d_in[4];   // encoded_graph_emb_p
    const float* Hg    = (const float*)d_in[5];   // encoded_graph_emb_h

    // d_out layout (flat concat, f32)
    float* out  = (float*)d_out;
    float* totp = out;
    float* toth = totp + ATT_N;
    float* aP   = toth + ATT_N;
    float* aPg  = aP   + TEN_N;
    float* aH   = aPg  + TEN_N;
    float* aHg  = aH   + TEN_N;
    float* ph   = aHg  + TEN_N;
    float* hp   = ph   + ATT_N;
    float* phg  = hp   + ATT_N;
    float* hpg  = phg  + ATT_N;

    // Workspace layout: [Tt x4][S,Sg][W x4][row-bf16 x4 (optional fast path)]
    char*   ws    = (char*)d_ws;
    size_t  szTt  = (size_t)TEN_N * sizeof(__bf16);
    size_t  szS   = (size_t)ATT_N * sizeof(float);
    size_t  szW   = (size_t)ATT_N * sizeof(__bf16);
    __bf16* TtH   = (__bf16*)(ws);
    __bf16* TtP   = (__bf16*)(ws + szTt);
    __bf16* TtHg  = (__bf16*)(ws + 2 * szTt);
    __bf16* TtPg  = (__bf16*)(ws + 3 * szTt);
    float*  S     = (float*)(ws + 4 * szTt);
    float*  Sg    = (float*)(ws + 4 * szTt + szS);
    __bf16* Wtotp = (__bf16*)(ws + 4 * szTt + 2 * szS);
    __bf16* Wtoth = Wtotp + ATT_N;
    __bf16* Wphg  = Wtoth + ATT_N;
    __bf16* Whpg  = Wphg  + ATT_N;
    size_t  base_need = 4 * szTt + 2 * szS + 4 * szW;        // ~66 MB
    size_t  full_need = base_need + 4 * szTt;                // ~114 MB
    bool    fast  = (ws_size >= full_need);
    __bf16* RH  = fast ? (__bf16*)(ws + base_need)            : TtH;  // dummies when !fast
    __bf16* RP  = fast ? (__bf16*)(ws + base_need + szTt)     : TtH;
    __bf16* RHg = fast ? (__bf16*)(ws + base_need + 2 * szTt) : TtH;
    __bf16* RPg = fast ? (__bf16*)(ws + base_need + 3 * szTt) : TtH;

    // 1) bf16 transposed (+ optional row-major) copies of the four tensors
    dim3 tb(32, 8);
    dim3 tg(DMODEL / 32, L / 32, BATCH * 4);
    transpose_cvt_kernel<<<tg, tb, 0, stream>>>(H, P, Hg, Pg,
                                                TtH, TtP, TtHg, TtPg,
                                                RH, RP, RHg, RPg,
                                                fast ? 1 : 0);

    // 2) similarity matrices via bf16 WMMA
    if (fast)
        sim_bf16_kernel<<<(2 * BATCH * 36) / 4, 128, 0, stream>>>(RP, RH, RPg, RHg, S, Sg);
    else
        sim_f32_kernel<<<(2 * BATCH * 144) / 4, 128, 0, stream>>>(P, H, Pg, Hg, S, Sg);

    // 3) masked softmaxes (row and column variants)
    masked_softmax_kernel<<<BATCH * L, 256, 0, stream>>>(S,  L, 1, maskH, ph);
    masked_softmax_kernel<<<BATCH * L, 256, 0, stream>>>(S,  1, L, maskP, hp);
    masked_softmax_kernel<<<BATCH * L, 256, 0, stream>>>(Sg, L, 1, maskH, phg);
    masked_softmax_kernel<<<BATCH * L, 256, 0, stream>>>(Sg, 1, L, maskP, hpg);

    // 4) tot_attn + bf16 weight copies
    combine_kernel<<<(ATT_N + 255) / 256, 256, 0, stream>>>(
        ph, hp, phg, hpg, totp, toth, Wtotp, Wtoth, Wphg, Whpg);

    // 5) weighted sums via bf16 WMMA (4-wide N blocking), masked at store
    wsum_kernel<<<(4 * BATCH * 192) / 4, 128, 0, stream>>>(
        Wtotp, Wtoth, Wphg, Whpg, TtH, TtP, TtHg, TtPg,
        maskP, maskH, aP, aH, aPg, aHg);
}
